// InteractionNetwork_85091892069039
// MI455X (gfx1250) — compile-verified, matrix operand order checked by reading, not run on hardware
//
#include <hip/hip_runtime.h>
#include <hip/hip_fp16.h>
#include <math.h>

// ---------------------------------------------------------------------------
// Fused InteractionNetwork for gfx1250 (MI455X).
// All GEMMs run through v_wmma_f32_16x16x32_f16 (f16 in, f32 accumulate).
// Weights are pre-converted once into padded/transposed f16 blobs (L2
// resident), bulk-copied into LDS per block; the 200-wide hidden
// activations live entirely in per-wave LDS/VGPRs. N-tiles are processed
// in pairs (two independent accumulator chains) to hide the WMMA->VALU
// hazard window and LDS latency.
// ---------------------------------------------------------------------------

typedef _Float16 f16;
typedef __attribute__((ext_vector_type(16))) _Float16 v16h;
typedef __attribute__((ext_vector_type(8)))  _Float16 v8h;
typedef __attribute__((ext_vector_type(8)))  float    v8f;

#define NFT    16    // node feature / latent width
#define HID    200   // mlp hidden size
#define NPAD   224   // hidden padded (14 N-tiles of 16, 7 K-chunks of 32)
#define NT     (NPAD / 16)   // 14
#define KC     (NPAD / 32)   // 7
#define KIN    64    // padded input width (covers 32/40/48)
#define WAVES  4
#define THREADS (WAVES * 32)
#define TILES  4     // 16-row tiles per wave -> 256 rows per block

// weight blob: [w1t NPAD*KIN | w2t NPAD*NPAD | w3t 16*NPAD] f16, then
// [b1 NPAD | b2 NPAD | b3 16] f32. LDS image is identical.
#define W1T_ELEMS (NPAD * KIN)       // 14336
#define W2T_ELEMS (NPAD * NPAD)      // 50176
#define W3T_ELEMS (16 * NPAD)        // 3584
#define BLOB_F16  (W1T_ELEMS + W2T_ELEMS + W3T_ELEMS)   // 68096
#define BLOB_F32  (2 * NPAD + 16)                       // 464
#define BLOB_BYTES (BLOB_F16 * 2 + BLOB_F32 * 4)        // 138048 (16B multiple)
#define ACT_F16_PER_WAVE (16 * KIN + 2 * 16 * NPAD)     // 8192
#define SMEM_BYTES (BLOB_BYTES + WAVES * ACT_F16_PER_WAVE * 2)

// Build a 16x32 f16 fragment per CDNA5 16-bit A-matrix layout:
// lane = row (lanes 0-15 / 16-31 split K halves), VGPR halves hold K groups.
// Same loader serves B by pointing `base` at the transposed weight tile row.
__device__ __forceinline__ v16h frag_a16(const f16* base, int stride, int k0, int lane) {
  const int row = lane & 15;
  const int hi  = (lane >> 4) & 1;
  const f16* p = base + row * stride + k0 + hi * 8;
  v16h v;
#pragma unroll
  for (int e = 0; e < 8; ++e) v[e] = p[e];
#pragma unroll
  for (int e = 0; e < 8; ++e) v[8 + e] = p[16 + e];
  return v;
}

// bias + ReLU + f16 pack + LDS store of one 16x16 D tile
__device__ __forceinline__ void store_tile_relu(f16* h, int col0, float bias,
                                                const v8f& acc, int lane) {
  const int nlo = lane & 15;
  const int hi  = (lane >> 4) & 1;
#pragma unroll
  for (int i = 0; i < 8; ++i) {
    float v = fmaxf(acc[i] + bias, 0.f);
    h[(hi * 8 + i) * NPAD + col0 + nlo] = (f16)v;
  }
}

// 3-layer MLP on one 16-row tile: [16,64] -> relu(224) -> relu(224) -> [16,16]
// (cols 200..223 are exact zeros via zero-padded weights/biases)
__device__ __forceinline__ void mlp3_64(
    const f16* inA, f16* h1, f16* h2,
    const f16* w1t, const float* b1,
    const f16* w2t, const float* b2,
    const f16* w3t, const float* b3,
    v8f& dout, int lane)
{
  const int nlo = lane & 15;

  // ---- Linear 1: [16,64] @ [64,224] + b1, ReLU -> h1 (paired N-tiles)
  {
    v16h a0 = frag_a16(inA, KIN, 0, lane);
    v16h a1 = frag_a16(inA, KIN, 32, lane);
    for (int j = 0; j < NT; j += 2) {
      const f16* w0 = w1t + (j * 16) * KIN;
      const f16* w1 = w1t + ((j + 1) * 16) * KIN;
      v8f acc0 = {}, acc1 = {};
      v16h bf;
      bf = frag_a16(w0, KIN, 0, lane);
      acc0 = __builtin_amdgcn_wmma_f32_16x16x32_f16(false, a0, false, bf, (short)0, acc0, false, false);
      bf = frag_a16(w1, KIN, 0, lane);
      acc1 = __builtin_amdgcn_wmma_f32_16x16x32_f16(false, a0, false, bf, (short)0, acc1, false, false);
      bf = frag_a16(w0, KIN, 32, lane);
      acc0 = __builtin_amdgcn_wmma_f32_16x16x32_f16(false, a1, false, bf, (short)0, acc0, false, false);
      bf = frag_a16(w1, KIN, 32, lane);
      acc1 = __builtin_amdgcn_wmma_f32_16x16x32_f16(false, a1, false, bf, (short)0, acc1, false, false);
      store_tile_relu(h1, j * 16, b1[j * 16 + nlo], acc0, lane);
      store_tile_relu(h1, (j + 1) * 16, b1[(j + 1) * 16 + nlo], acc1, lane);
    }
  }

  // ---- Linear 2: [16,224] @ [224,224] + b2, ReLU -> h2 (paired N-tiles)
  {
    v16h a[KC];
#pragma unroll
    for (int kc = 0; kc < KC; ++kc)
      a[kc] = frag_a16(h1, NPAD, kc * 32, lane);
    for (int j = 0; j < NT; j += 2) {
      const f16* w0 = w2t + (j * 16) * NPAD;
      const f16* w1 = w2t + ((j + 1) * 16) * NPAD;
      v8f acc0 = {}, acc1 = {};
#pragma unroll
      for (int kc = 0; kc < KC; ++kc) {
        v16h bf0 = frag_a16(w0, NPAD, kc * 32, lane);
        acc0 = __builtin_amdgcn_wmma_f32_16x16x32_f16(false, a[kc], false, bf0, (short)0, acc0, false, false);
        v16h bf1 = frag_a16(w1, NPAD, kc * 32, lane);
        acc1 = __builtin_amdgcn_wmma_f32_16x16x32_f16(false, a[kc], false, bf1, (short)0, acc1, false, false);
      }
      store_tile_relu(h2, j * 16, b2[j * 16 + nlo], acc0, lane);
      store_tile_relu(h2, (j + 1) * 16, b2[(j + 1) * 16 + nlo], acc1, lane);
    }
  }

  // ---- Linear 3: [16,224] @ [224,16] + b3 (no activation)
  {
    v8f acc = {};
#pragma unroll
    for (int kc = 0; kc < KC; ++kc) {
      v16h a  = frag_a16(h2, NPAD, kc * 32, lane);
      v16h bf = frag_a16(w3t, NPAD, kc * 32, lane);
      acc = __builtin_amdgcn_wmma_f32_16x16x32_f16(false, a, false, bf, (short)0, acc, false, false);
    }
    float bias = b3[nlo];
#pragma unroll
    for (int i = 0; i < 8; ++i) acc[i] += bias;
    dout = acc;
  }
}

// Bulk-copy the pre-formatted weight blob straight into LDS (16B vectors).
__device__ __forceinline__ void stage_blob(const char* blob, char* smem) {
  const int tid = threadIdx.x, nth = blockDim.x;
  __builtin_prefetch(blob + tid * 16, 0, 0);     // global_prefetch_b8
  const uint4* s = (const uint4*)blob;
  uint4* d = (uint4*)smem;
  for (int t = tid; t < BLOB_BYTES / 16; t += nth) d[t] = s[t];
  __syncthreads();
}

// pack 32 gathered floats into 4 x 16B LDS stores
__device__ __forceinline__ void pack_store32(f16* out, const float* vals) {
#pragma unroll
  for (int g = 0; g < 4; ++g) {
    v8h h;
#pragma unroll
    for (int u = 0; u < 8; ++u) h[u] = (f16)vals[g * 8 + u];
    *(v8h*)(out + g * 8) = h;
  }
}

// One-time weight prep: transpose+pad W to f16 [N][K] blobs, pad biases.
__global__ void prep_weights(const float* __restrict__ W1, const float* __restrict__ B1,
                             const float* __restrict__ W2, const float* __restrict__ B2,
                             const float* __restrict__ W3, const float* __restrict__ B3,
                             int in_dim, int out3,
                             f16* __restrict__ wb, float* __restrict__ bb)
{
  int i = blockIdx.x * blockDim.x + threadIdx.x;
  if (i < W1T_ELEMS) {
    int n = i / KIN, k = i % KIN;
    wb[i] = (f16)((k < in_dim && n < HID) ? W1[k * HID + n] : 0.f);
  } else if (i < W1T_ELEMS + W2T_ELEMS) {
    int t = i - W1T_ELEMS, n = t / NPAD, k = t % NPAD;
    wb[i] = (f16)((k < HID && n < HID) ? W2[k * HID + n] : 0.f);
  } else if (i < BLOB_F16) {
    int t = i - W1T_ELEMS - W2T_ELEMS, n = t / NPAD, k = t % NPAD;
    wb[i] = (f16)((k < HID && n < out3) ? W3[k * out3 + n] : 0.f);
  } else if (i < BLOB_F16 + BLOB_F32) {
    int t = i - BLOB_F16;
    float v;
    if (t < NPAD)            v = (t < HID) ? B1[t] : 0.f;
    else if (t < 2 * NPAD)   { int u = t - NPAD;     v = (u < HID)  ? B2[u] : 0.f; }
    else                     { int u = t - 2 * NPAD; v = (u < out3) ? B3[u] : 0.f; }
    bb[t] = v;
  }
}

// ---------------------------------------------------------------------------
// Edge-side kernel: gather [x[dst], x[src], ea] -> fused MLP -> msg (+atomic
// segment-sum scatter into aggr) or sigmoid classifier output.
// ---------------------------------------------------------------------------
template<int EF_, int OUT3, bool SCATTER, bool SIGMOID>
__global__ void edge_mlp_kernel(
    const float* __restrict__ x, const int* __restrict__ ei,
    const float* __restrict__ ea, const char* __restrict__ blob,
    float* __restrict__ out, float* __restrict__ aggr, int E)
{
  extern __shared__ char smem[];
  const f16*   w1t = (const f16*)smem;
  const f16*   w2t = w1t + W1T_ELEMS;
  const f16*   w3t = w2t + W2T_ELEMS;
  const float* b1  = (const float*)(smem + BLOB_F16 * 2);
  const float* b2  = b1 + NPAD;
  const float* b3  = b2 + NPAD;
  f16* acts = (f16*)(smem + BLOB_BYTES);

  const int wave = threadIdx.x >> 5;
  const int lane = threadIdx.x & 31;
  f16* inA = acts + wave * ACT_F16_PER_WAVE;
  f16* h1  = inA + 16 * KIN;
  f16* h2  = h1 + 16 * NPAD;

  stage_blob(blob, smem);

  for (int t = 0; t < TILES; ++t) {
    const long long base =
        ((long long)blockIdx.x * WAVES * TILES + wave * TILES + t) * 16;

    // gather: lane = row; lane-half selects 32-column half; float4 loads
    {
      const int r = lane & 15;
      const int half = lane >> 4;
      const long long e = base + r;
      const bool ok = (e < E);
      float vals[32];
#pragma unroll
      for (int i = 0; i < 32; ++i) vals[i] = 0.f;
      if (ok) {
        if (half == 0) {   // cols 0..31: x[dst] | x[src]
          const float4* pd = (const float4*)(x + (long long)ei[E + e] * NFT);
          const float4* ps = (const float4*)(x + (long long)ei[e] * NFT);
#pragma unroll
          for (int q = 0; q < 4; ++q) {
            float4 f = pd[q];
            vals[q*4+0]=f.x; vals[q*4+1]=f.y; vals[q*4+2]=f.z; vals[q*4+3]=f.w;
          }
#pragma unroll
          for (int q = 0; q < 4; ++q) {
            float4 f = ps[q];
            vals[16+q*4+0]=f.x; vals[16+q*4+1]=f.y; vals[16+q*4+2]=f.z; vals[16+q*4+3]=f.w;
          }
        } else {           // cols 32..63: edge_attr | 0-pad
          const float4* pe = (const float4*)(ea + e * EF_);
#pragma unroll
          for (int q = 0; q < EF_ / 4; ++q) {
            float4 f = pe[q];
            vals[q*4+0]=f.x; vals[q*4+1]=f.y; vals[q*4+2]=f.z; vals[q*4+3]=f.w;
          }
        }
      }
      pack_store32(inA + r * KIN + half * 32, vals);
    }

    v8f d;
    mlp3_64(inA, h1, h2, w1t, b1, w2t, b2, w3t, b3, d, lane);

    const int nlo = lane & 15, hi = (lane >> 4) & 1;
#pragma unroll
    for (int i = 0; i < 8; ++i) {
      long long e = base + hi * 8 + i;
      if (e >= E) continue;
      float v = d[i];
      if (SIGMOID) {
        if (nlo == 0) out[e] = 1.f / (1.f + __expf(-v));
      } else if (nlo < OUT3) {
        out[e * OUT3 + nlo] = v;
        if (SCATTER) {
          int dst = ei[E + e];
          atomicAdd(&aggr[(long long)dst * OUT3 + nlo], v);
        }
      }
    }
  }
}

// ---------------------------------------------------------------------------
// Node-side update kernel: [x | aggr | 0-pad] (64) -> fused MLP -> x' (16)
// ---------------------------------------------------------------------------
__global__ void node_mlp_kernel(
    const float* __restrict__ x, const float* __restrict__ aggr,
    const char* __restrict__ blob, float* __restrict__ xout, int N)
{
  extern __shared__ char smem[];
  const f16*   w1t = (const f16*)smem;
  const f16*   w2t = w1t + W1T_ELEMS;
  const f16*   w3t = w2t + W2T_ELEMS;
  const float* b1  = (const float*)(smem + BLOB_F16 * 2);
  const float* b2  = b1 + NPAD;
  const float* b3  = b2 + NPAD;
  f16* acts = (f16*)(smem + BLOB_BYTES);

  const int wave = threadIdx.x >> 5;
  const int lane = threadIdx.x & 31;
  f16* inA = acts + wave * ACT_F16_PER_WAVE;
  f16* h1  = inA + 16 * KIN;
  f16* h2  = h1 + 16 * NPAD;

  stage_blob(blob, smem);

  for (int t = 0; t < TILES; ++t) {
    const long long base =
        ((long long)blockIdx.x * WAVES * TILES + wave * TILES + t) * 16;
    {
      const int r = lane & 15;
      const int half = lane >> 4;
      const long long n = base + r;
      float vals[32];
#pragma unroll
      for (int i = 0; i < 32; ++i) vals[i] = 0.f;
      if (n < N && half == 0) {   // cols 0..31: x | aggr (cols 32..63 stay 0)
        const float4* px = (const float4*)(x + n * NFT);
        const float4* pa = (const float4*)(aggr + n * NFT);
#pragma unroll
        for (int q = 0; q < 4; ++q) {
          float4 f = px[q];
          vals[q*4+0]=f.x; vals[q*4+1]=f.y; vals[q*4+2]=f.z; vals[q*4+3]=f.w;
        }
#pragma unroll
        for (int q = 0; q < 4; ++q) {
          float4 f = pa[q];
          vals[16+q*4+0]=f.x; vals[16+q*4+1]=f.y; vals[16+q*4+2]=f.z; vals[16+q*4+3]=f.w;
        }
      }
      pack_store32(inA + r * KIN + half * 32, vals);
    }

    v8f d;
    mlp3_64(inA, h1, h2, w1t, b1, w2t, b2, w3t, b3, d, lane);

    const int nlo = lane & 15, hi = (lane >> 4) & 1;
#pragma unroll
    for (int i = 0; i < 8; ++i) {
      long long n = base + hi * 8 + i;
      if (n < N) xout[n * 16 + nlo] = d[i];
    }
  }
}

__global__ void zero_kernel(float* __restrict__ p, long long n) {
  long long i = (long long)blockIdx.x * blockDim.x + threadIdx.x;
  if (i < n) p[i] = 0.f;
}

extern "C" void kernel_launch(void* const* d_in, const int* in_sizes, int n_in,
                              void* d_out, int out_size, void* d_ws, size_t ws_size,
                              hipStream_t stream)
{
  (void)n_in; (void)out_size; (void)ws_size;
  const float* node_attr = (const float*)d_in[0];
  const int*   ei        = (const int*)d_in[1];   // [2,E]: row0=src, row1=dst
  const float* edge_attr = (const float*)d_in[2];

  // params pytree (dict keys sorted): clf first, then layers[0].msg,
  // layers[0].upd, layers[1].msg, layers[1].upd; each MLP = 3x (W, b).
  const float* const* P = (const float* const*)d_in;

  const int N = in_sizes[0] / NFT;
  const int E = in_sizes[2] / 8;

  float* aggr = (float*)d_ws;
  float* x1   = aggr + (size_t)N * 16;
  float* x2   = x1   + (size_t)N * 16;
  float* msg0 = x2   + (size_t)N * 16;
  float* msg1 = msg0 + (size_t)E * 16;
  char*  blobs = (char*)(msg1 + (size_t)E * 16);
  // blob order: 0=msg0, 1=upd0, 2=msg1, 3=upd1, 4=clf
  char* bl[5];
  for (int i = 0; i < 5; ++i) bl[i] = blobs + (size_t)i * BLOB_BYTES;

  const dim3 blk(THREADS);
  const int rows_pb = WAVES * TILES * 16;                 // 256
  const int eb = (E + rows_pb - 1) / rows_pb;
  const int nb = (N + rows_pb - 1) / rows_pb;
  const long long an = (long long)N * 16;
  const int zb = (int)((an + 255) / 256);
  const int pb = (BLOB_F16 + BLOB_F32 + 255) / 256;

  // ---- one-time weight prep (params: idx 3.. in pytree order)
  struct MSpec { int pi; int in_dim; int out3; int blob; };
  const MSpec specs[5] = {
      {9,  40, 16, 0},   // layers[0].msg
      {15, 32, 16, 1},   // layers[0].upd
      {21, 48, 16, 2},   // layers[1].msg
      {27, 32, 16, 3},   // layers[1].upd
      {3,  48, 1,  4},   // clf
  };
  for (int s = 0; s < 5; ++s) {
    const int p = specs[s].pi;
    prep_weights<<<pb, 256, 0, stream>>>(
        P[p], P[p + 1], P[p + 2], P[p + 3], P[p + 4], P[p + 5],
        specs[s].in_dim, specs[s].out3,
        (f16*)bl[specs[s].blob], (float*)(bl[specs[s].blob] + BLOB_F16 * 2));
  }

  // ---- layer 0
  zero_kernel<<<zb, 256, 0, stream>>>(aggr, an);
  edge_mlp_kernel<8, 16, true, false><<<eb, blk, SMEM_BYTES, stream>>>(
      node_attr, ei, edge_attr, bl[0], msg0, aggr, E);
  node_mlp_kernel<<<nb, blk, SMEM_BYTES, stream>>>(node_attr, aggr, bl[1], x1, N);

  // ---- layer 1
  zero_kernel<<<zb, 256, 0, stream>>>(aggr, an);
  edge_mlp_kernel<16, 16, true, false><<<eb, blk, SMEM_BYTES, stream>>>(
      x1, ei, msg0, bl[2], msg1, aggr, E);
  node_mlp_kernel<<<nb, blk, SMEM_BYTES, stream>>>(x1, aggr, bl[3], x2, N);

  // ---- edge classifier + sigmoid
  edge_mlp_kernel<16, 1, false, true><<<eb, blk, SMEM_BYTES, stream>>>(
      x2, ei, msg1, bl[4], (float*)d_out, nullptr, E);
}
